// ImportanceRenderer_51651276701981
// MI455X (gfx1250) — compile-verified, matrix operand order checked by reading, not run on hardware
//
#include <hip/hip_runtime.h>

typedef __attribute__((ext_vector_type(16))) _Float16 v16h;
typedef __attribute__((ext_vector_type(8)))  float    v8f;

#define RAY_START 2.25f
#define RAY_END   3.3f
#define SD   48     // coarse depth res
#define NI   48     // n importance
#define CH   32     // plane channels
#define HW   256    // plane H=W
#define NTOT 96     // merged samples
#define NTHREADS 96

struct SharedBlk {
  alignas(16) float wsh1[CH * 64];   // dec_w1 staged via async DMA (32x64)
  alignas(16) float wsh2[64 * 33];   // dec_w2 staged via async DMA (64x33)
  _Float16 h[3][16][64];     // per-wave hidden activations (f16 for layer-2 A frags)
  float    o[2][SD][33];     // [phase][sample][0]=sigma, [1..32]=rgb(post-sigmoid)
  float    zfine[NI];
  float    cdf[46];
  float    dall[NTOT];
  int      src[NTOT];
  float    wts[NTOT - 1];
  float    hbg[64];
  float    cbg[32];
  float    ray_o[3];
  float    ray_d[3];
  float    bgT;
};

__device__ __forceinline__ int kmap(int e, int halfSel) {
  // element e of a v16h A/B fragment -> K index (16-bit 16x32 layout, ISA 7.12.2)
  return ((e >> 3) << 4) + (halfSel << 3) + (e & 7);
}

// fast softplus: max(x,0) + log(1+exp(-|x|)) using v_exp_f32/v_log_f32
__device__ __forceinline__ float softplus_f(float x) {
  return fmaxf(x, 0.f) + __logf(1.f + __expf(-fabsf(x)));
}
__device__ __forceinline__ float sigmoid_f(float x) {
  return 1.f / (1.f + __expf(-x));
}

// Async DMA: global f32 array -> LDS, 16B granules, ASYNCcnt-tracked.
__device__ __forceinline__ void async_stage_f32(const float* gsrc, float* lds_dst,
                                                int nfloat, int tid) {
  const int nvec = nfloat >> 2;  // b128 chunks
  for (int i = tid; i < nvec; i += NTHREADS) {
    // ISA 10.2: flat shared addr low 32 bits == LDS byte offset
    unsigned loff = (unsigned)(size_t)(lds_dst + i * 4);
    unsigned long long ga = (unsigned long long)(size_t)(gsrc + i * 4);
    asm volatile("global_load_async_to_lds_b128 %0, %1, off"
                 :: "v"(loff), "v"(ga) : "memory");
  }
}

// One wave decodes 16 samples: tri-plane gather -> MLP(32->64->33) via WMMA.
// w1s/w2s point into LDS (staged weights).
__device__ void decode16(const float* planesB, const float* w1s, const float* b1,
                         const float* w2s, const float* b2, SharedBlk* sm,
                         int phase, int wv, int lane,
                         float ox, float oy, float oz,
                         float dx, float dy, float dz) {
  const int   halfSel = lane >> 4;
  const int   nm      = lane & 15;      // sample-row (A) / col-N (B,C)
  const int   s       = wv * 16 + nm;   // sample 0..47
  const float delta   = (RAY_END - RAY_START) / (SD - 1);

  float depth = (phase == 0) ? (RAY_START + ((float)s + 0.5f) * delta)
                             : sm->zfine[s];
  // c = (2/box_warp) * world
  const float cx = 2.f * (ox + depth * dx);
  const float cy = 2.f * (oy + depth * dy);
  const float cz = 2.f * (oz + depth * dz);

  float feats[16];
#pragma unroll
  for (int e = 0; e < 16; ++e) feats[e] = 0.f;

#pragma unroll
  for (int p = 0; p < 3; ++p) {
    float gx, gy;
    if      (p == 0) { gx = cx; gy = cy; }   // (x,y)
    else if (p == 1) { gx = cx; gy = cz; }   // (x,z)
    else             { gx = cz; gy = cx; }   // (z,x)
    float x = gx * (HW * 0.5f) + (HW * 0.5f - 0.5f);
    float y = gy * (HW * 0.5f) + (HW * 0.5f - 0.5f);
    float x0f = floorf(x), y0f = floorf(y);
    float fx = x - x0f, fy = y - y0f;
    int x0 = (int)x0f, y0 = (int)y0f, x1 = x0 + 1, y1 = y0 + 1;
    float vx0 = (x0 >= 0 && x0 < HW) ? 1.f : 0.f;
    float vx1 = (x1 >= 0 && x1 < HW) ? 1.f : 0.f;
    float vy0 = (y0 >= 0 && y0 < HW) ? 1.f : 0.f;
    float vy1 = (y1 >= 0 && y1 < HW) ? 1.f : 0.f;
    int cx0 = min(max(x0, 0), HW - 1), cx1 = min(max(x1, 0), HW - 1);
    int cy0 = min(max(y0, 0), HW - 1), cy1 = min(max(y1, 0), HW - 1);
    float w00 = (1.f - fx) * (1.f - fy) * vx0 * vy0;
    float w10 = fx * (1.f - fy) * vx1 * vy0;
    float w01 = (1.f - fx) * fy * vx0 * vy1;
    float w11 = fx * fy * vx1 * vy1;
    const float* pb = planesB + (size_t)p * CH * HW * HW;
    int i00 = cy0 * HW + cx0, i10 = cy0 * HW + cx1;
    int i01 = cy1 * HW + cx0, i11 = cy1 * HW + cx1;
#pragma unroll
    for (int e = 0; e < 16; ++e) {
      const float* fp = pb + (size_t)kmap(e, halfSel) * (HW * HW);
      feats[e] += w00 * fp[i00] + w10 * fp[i10] + w01 * fp[i01] + w11 * fp[i11];
    }
  }

  v16h a;
#pragma unroll
  for (int e = 0; e < 16; ++e) a[e] = (_Float16)(feats[e] * (1.f / 3.f));

  // layer 1: (16x32)@(32x64) -> four 16x16 N-tiles
#pragma unroll
  for (int t = 0; t < 4; ++t) {
    v16h bw;
#pragma unroll
    for (int e = 0; e < 16; ++e)
      bw[e] = (_Float16)w1s[kmap(e, halfSel) * 64 + t * 16 + nm];
    v8f c = {};
    c = __builtin_amdgcn_wmma_f32_16x16x32_f16(false, a, false, bw, (short)0, c,
                                               false, false);
#pragma unroll
    for (int v = 0; v < 8; ++v) {
      int row = v + halfSel * 8;
      int col = t * 16 + nm;
      sm->h[wv][row][col] = (_Float16)softplus_f(c[v] + b1[col]);
    }
  }
  __syncthreads();

  // layer 2: (16x64)@(64x33, padded to 48)
  v16h a2[2];
#pragma unroll
  for (int kt = 0; kt < 2; ++kt)
#pragma unroll
    for (int e = 0; e < 16; ++e)
      a2[kt][e] = sm->h[wv][nm][kt * 32 + kmap(e, halfSel)];

#pragma unroll
  for (int nt = 0; nt < 3; ++nt) {
    v8f c = {};
#pragma unroll
    for (int kt = 0; kt < 2; ++kt) {
      v16h bw;
#pragma unroll
      for (int e = 0; e < 16; ++e) {
        int k   = kt * 32 + kmap(e, halfSel);
        int col = nt * 16 + nm;
        bw[e] = (col < 33) ? (_Float16)w2s[k * 33 + col] : (_Float16)0.f;
      }
      c = __builtin_amdgcn_wmma_f32_16x16x32_f16(false, a2[kt], false, bw,
                                                 (short)0, c, false, false);
    }
    int col = nt * 16 + nm;
    if (col < 33) {
#pragma unroll
      for (int v = 0; v < 8; ++v) {
        int   row = v + halfSel * 8;
        int   ss  = wv * 16 + row;
        float val = c[v] + b2[col];
        if (col == 0) sm->o[phase][ss][0] = val;                       // sigma raw
        else sm->o[phase][ss][col] = sigmoid_f(val) * 1.002f - 0.001f; // rgb
      }
    }
  }
  __syncthreads();
}

__global__ __launch_bounds__(NTHREADS) void importance_renderer_kernel(
    const float* planes, const float* ray_o, const float* ray_d,
    const float* z_bg, const float* w1, const float* b1, const float* w2,
    const float* b2, const float* bw1, const float* bb1, const float* bw2,
    const float* bb2, float* out, int B, int R) {
  __shared__ SharedBlk sm;
  const int tid  = threadIdx.x;
  const int wv   = tid >> 5;
  const int lane = tid & 31;
  const int ridx = blockIdx.x;   // 0..B*R-1
  const int b    = ridx / R;
  const float delta = (RAY_END - RAY_START) / (SD - 1);

  // kick off async weight staging (overlaps with bg MLP below)
  async_stage_f32(w1, sm.wsh1, CH * 64, tid);
  async_stage_f32(w2, sm.wsh2, 64 * 33, tid);

  if (tid < 3) {
    sm.ray_o[tid] = ray_o[(size_t)ridx * 3 + tid];
    sm.ray_d[tid] = ray_d[(size_t)ridx * 3 + tid];
  }
  __syncthreads();
  const float ox = sm.ray_o[0], oy = sm.ray_o[1], oz = sm.ray_o[2];
  const float dx = sm.ray_d[0], dy = sm.ray_d[1], dz = sm.ray_d[2];
  const float* planesB = planes + (size_t)b * 3 * CH * HW * HW;

  // ---- background MLP (67->64 softplus ->32 sigmoid) ----
  if (tid < 64) {
    float acc = bb1[tid];
    acc += dx * bw1[0 * 64 + tid] + dy * bw1[1 * 64 + tid] + dz * bw1[2 * 64 + tid];
    const float* zb = z_bg + (size_t)b * 64;
    for (int i = 0; i < 64; ++i) acc += zb[i] * bw1[(3 + i) * 64 + tid];
    sm.hbg[tid] = softplus_f(acc);
  }
  __syncthreads();
  if (tid < 32) {
    float acc = bb2[tid];
    for (int j = 0; j < 64; ++j) acc += sm.hbg[j] * bw2[j * 32 + tid];
    sm.cbg[tid] = sigmoid_f(acc);
  }

  // fence the weight DMA (per-wave count), then make visible block-wide
  asm volatile("s_wait_asynccnt 0x0" ::: "memory");
  __syncthreads();

  // ---- coarse pass ----
  decode16(planesB, sm.wsh1, b1, sm.wsh2, b2, &sm, 0, wv, lane,
           ox, oy, oz, dx, dy, dz);

  // ---- coarse march + importance CDF (thread 0) ----
  if (tid == 0) {
    float wc[SD - 1];
    float T = 1.f;
    for (int i = 0; i < SD - 1; ++i) {
      float dmid  = softplus_f(0.5f * (sm.o[0][i][0] + sm.o[0][i + 1][0]) - 1.f);
      float alpha = 1.f - __expf(-dmid * delta);
      wc[i] = alpha * T;
      T *= (1.f - alpha + 1e-10f);
    }
    float w[SD - 1];
    for (int i = 0; i < SD - 1; ++i) w[i] = wc[i] + 0.01f;
    float wmax[SD];
    for (int i = 0; i < SD; ++i) {
      float a = (i >= 1)     ? w[i - 1] : -__builtin_inff();
      float c = (i < SD - 1) ? w[i]     : -__builtin_inff();
      wmax[i] = fmaxf(a, c);
    }
    float w2a[SD - 1];
    for (int i = 0; i < SD - 1; ++i) w2a[i] = 0.5f * (wmax[i] + wmax[i + 1]) + 0.01f;
    // pdf over w2a[1..45]  (45 weights, 47 bins at z_mid)
    float tot = 0.f;
    for (int i = 1; i <= 45; ++i) tot += (w2a[i] + 1e-5f);
    float cum = 0.f;
    sm.cdf[0] = 0.f;
    for (int i = 1; i <= 45; ++i) {
      cum += (w2a[i] + 1e-5f) / tot;
      sm.cdf[i] = cum;
    }
  }
  __syncthreads();

  // ---- inverse-CDF sampling (48 threads) ----
  if (tid < NI) {
    float u = ((float)tid + 0.5f) / (float)NI;
    int ind = 0;
    while (ind < 46 && sm.cdf[ind] <= u) ++ind;   // searchsorted 'right'
    int below = ind - 1;
    below = below < 0 ? 0 : (below > 45 ? 45 : below);
    int above = ind > 45 ? 45 : ind;
    float cb = sm.cdf[below], ca = sm.cdf[above];
    float bb = RAY_START + (float)(below + 1) * delta;   // z_mid[below]
    float ba = RAY_START + (float)(above + 1) * delta;   // z_mid[above]
    float den = (ca - cb < 1e-5f) ? 1.f : (ca - cb);
    sm.zfine[tid] = bb + (u - cb) / den * (ba - bb);
  }
  __syncthreads();

  // ---- fine pass ----
  decode16(planesB, sm.wsh1, b1, sm.wsh2, b2, &sm, 1, wv, lane,
           ox, oy, oz, dx, dy, dz);

  // ---- merge (stable, both sorted) + final march scalars (thread 0) ----
  if (tid == 0) {
    int i = 0, j = 0;
    for (int k = 0; k < NTOT; ++k) {
      float dc = (i < SD) ? (RAY_START + ((float)i + 0.5f) * delta) : __builtin_inff();
      float df = (j < NI) ? sm.zfine[j] : __builtin_inff();
      if (dc <= df) { sm.dall[k] = dc; sm.src[k] = i;      ++i; }
      else          { sm.dall[k] = df; sm.src[k] = SD + j; ++j; }
    }
    float T = 1.f, wtot = 0.f, dsum = 0.f;
    for (int k = 0; k < NTOT - 1; ++k) {
      int s0 = sm.src[k], s1 = sm.src[k + 1];
      float sg0 = (s0 < SD) ? sm.o[0][s0][0] : sm.o[1][s0 - SD][0];
      float sg1 = (s1 < SD) ? sm.o[0][s1][0] : sm.o[1][s1 - SD][0];
      float dmid  = softplus_f(0.5f * (sg0 + sg1) - 1.f);
      float dd    = sm.dall[k + 1] - sm.dall[k];
      float alpha = 1.f - __expf(-dmid * dd);
      float wgt   = alpha * T;
      sm.wts[k] = wgt;
      wtot += wgt;
      dsum += wgt * 0.5f * (sm.dall[k] + sm.dall[k + 1]);
      T *= (1.f - alpha + 1e-10f);
    }
    sm.bgT = T;
    float cd = dsum / wtot;
    if (!(cd == cd)) cd = __builtin_inff();            // nan -> +inf
    cd = fminf(fmaxf(cd, sm.dall[0]), sm.dall[NTOT - 1]);
    size_t nray = (size_t)B * (size_t)R;
    out[nray * 32 + ridx] = cd;     // depth
    out[nray * 33 + ridx] = wtot;   // weights.sum
  }
  __syncthreads();

  // ---- composited rgb (32 threads, one per channel) ----
  if (tid < 32) {
    float acc = 0.f;
    for (int k = 0; k < NTOT - 1; ++k) {
      int s0 = sm.src[k], s1 = sm.src[k + 1];
      float c0 = (s0 < SD) ? sm.o[0][s0][1 + tid] : sm.o[1][s0 - SD][1 + tid];
      float c1 = (s1 < SD) ? sm.o[0][s1][1 + tid] : sm.o[1][s1 - SD][1 + tid];
      acc += sm.wts[k] * 0.5f * (c0 + c1);
    }
    acc += sm.bgT * sm.cbg[tid];
    out[(size_t)ridx * 32 + tid] = acc * 2.f - 1.f;
  }
}

extern "C" void kernel_launch(void* const* d_in, const int* in_sizes, int n_in,
                              void* d_out, int out_size, void* d_ws,
                              size_t ws_size, hipStream_t stream) {
  (void)n_in; (void)out_size; (void)d_ws; (void)ws_size;
  const float* planes = (const float*)d_in[0];
  const float* ray_o  = (const float*)d_in[1];
  const float* ray_d  = (const float*)d_in[2];
  const float* z_bg   = (const float*)d_in[3];
  const float* w1     = (const float*)d_in[4];
  const float* b1     = (const float*)d_in[5];
  const float* w2     = (const float*)d_in[6];
  const float* b2     = (const float*)d_in[7];
  const float* bw1    = (const float*)d_in[8];
  const float* bb1    = (const float*)d_in[9];
  const float* bw2    = (const float*)d_in[10];
  const float* bb2    = (const float*)d_in[11];

  int B = in_sizes[0] / (3 * CH * HW * HW);   // 4
  int R = in_sizes[1] / (3 * B);              // 4096

  dim3 grid((unsigned)(B * R)), block(NTHREADS);
  importance_renderer_kernel<<<grid, block, 0, stream>>>(
      planes, ray_o, ray_d, z_bg, w1, b1, w2, b2, bw1, bb1, bw2, bb2,
      (float*)d_out, B, R);
}
